// Attention_2087354105893
// MI455X (gfx1250) — compile-verified, matrix operand order checked by reading, not run on hardware
//
#include <hip/hip_runtime.h>
#include <hip/hip_bf16.h>

#define DEVFN static __device__ __forceinline__

#ifndef __has_builtin
#define __has_builtin(x) 0
#endif

#if __has_builtin(__builtin_amdgcn_global_load_async_to_lds_b128)
#define HAVE_ASYNC 1
#else
#define HAVE_ASYNC 0
#endif

#if HAVE_ASYNC
#if __has_builtin(__builtin_amdgcn_s_wait_asynccnt)
#define WAIT_ASYNC(n) __builtin_amdgcn_s_wait_asynccnt(n)
#else
#define WAIT_ASYNC(n) asm volatile("s_wait_asynccnt " #n ::: "memory")
#endif
#define AS_GLOBAL __attribute__((address_space(1)))
#define AS_LDS    __attribute__((address_space(3)))
// Builtin signature: (int4 AS1* src, int4 AS3* dst, imm offset, imm cpol)
typedef __attribute__((__vector_size__(16))) int v4i_t;
#endif

constexpr int BATCH = 8;
constexpr int CH    = 256;
constexpr int IH    = 56;
constexpr int IW    = 56;
constexpr int NQ    = IH * IW;      // 3136
constexpr int HKk   = IH / 2;       // 28
constexpr int WKk   = IW / 2;       // 28
constexpr int NK    = HKk * WKk;    // 784
constexpr int NKP   = 800;          // padded key count for v^T rows (16B-load safe)
constexpr int HEADS = 8;
constexpr int HD    = 32;
constexpr float ATT_SCALE = 0.17677669529663687f;  // 32^-0.5

// ---- workspace layout (bytes) ----
constexpr size_t SZ_RPE = (size_t)HEADS * NQ * NK * 2;        // 39,337,984
constexpr size_t SZ_XT  = (size_t)BATCH * NQ * CH * 2;        // 12,845,056
constexpr size_t SZ_WQ  = (size_t)CH * CH * 2;                //    131,072
constexpr size_t SZ_WKV = (size_t)2 * CH * CH * 2;            //    262,144
constexpr size_t SZ_T   = (size_t)BATCH * CH * NK * 4;        //  6,422,528
constexpr size_t SZ_KVT = (size_t)BATCH * NK * CH * 2;        //  3,211,264
constexpr size_t SZ_Q   = (size_t)BATCH * HEADS * NQ * HD * 2;// 12,845,056
constexpr size_t SZ_K   = (size_t)BATCH * HEADS * NK * HD * 2;//  3,211,264
constexpr size_t OFF_RPE = 0;
constexpr size_t OFF_XT  = OFF_RPE + SZ_RPE;
constexpr size_t OFF_WQ  = OFF_XT + SZ_XT;
constexpr size_t OFF_WKV = OFF_WQ + SZ_WQ;
constexpr size_t OFF_T   = OFF_WKV + SZ_WKV;
constexpr size_t OFF_KVT = OFF_T + SZ_T;
constexpr size_t OFF_Q   = OFF_KVT + SZ_KVT;
constexpr size_t OFF_K   = OFF_Q + SZ_Q;
constexpr size_t OFF_VT  = OFF_K + SZ_K;

// ---- types ----
typedef __attribute__((ext_vector_type(16))) __bf16          v16bf;
typedef __attribute__((ext_vector_type(16))) unsigned short  v16u;
typedef __attribute__((ext_vector_type(8)))  unsigned short  v8u;
typedef __attribute__((ext_vector_type(4)))  unsigned short  v4u;
typedef __attribute__((ext_vector_type(8)))  float           v8f;

DEVFN float bf2f(unsigned short h) { return __uint_as_float(((unsigned)h) << 16); }
DEVFN unsigned short f2bf(float f) {
  unsigned u = __float_as_uint(f);
  u += 0x7FFFu + ((u >> 16) & 1u);
  return (unsigned short)(u >> 16);
}

// Load one 16-bit WMMA A/B operand for this lane.
// 16-bit operand swizzle: lane holds halves k in [kb, kb+8) and [kb+16, kb+24),
// kb = 0 for lanes 0..15, 8 for lanes 16..31. Row base `p` must be 16B aligned.
DEVFN v16bf ldop(const unsigned short* p, int kb) {
  union { v16u u; v16bf b; } t;
  *reinterpret_cast<v8u*>(&t.u)       = *reinterpret_cast<const v8u*>(p + kb);
  *(reinterpret_cast<v8u*>(&t.u) + 1) = *reinterpret_cast<const v8u*>(p + kb + 16);
  return t.b;
}

DEVFN v8f wmma_bf16(v16bf a, v16bf b, v8f c) {
  return __builtin_amdgcn_wmma_f32_16x16x32_bf16(false, a, false, b, (short)0, c,
                                                 false, false);
}

// ---------------- prep kernels ----------------
__global__ __launch_bounds__(256) void k_prep_rpe(const float4* __restrict__ in,
                                                  v4u* __restrict__ out, int n4) {
  int i = blockIdx.x * 256 + threadIdx.x;
  if (i >= n4) return;
  float4 f = in[i];
  v4u o;
  o[0] = f2bf(f.x); o[1] = f2bf(f.y); o[2] = f2bf(f.z); o[3] = f2bf(f.w);
  out[i] = o;
}

__global__ __launch_bounds__(256) void k_prep_w(const float* __restrict__ qw,
                                                const float* __restrict__ kvw,
                                                unsigned short* __restrict__ wq,
                                                unsigned short* __restrict__ wkv) {
  int i = blockIdx.x * 256 + threadIdx.x;
  if (i < CH * CH) {
    wq[i] = f2bf(qw[i]);
  } else {
    int j = i - CH * CH;
    if (j < 2 * CH * CH) wkv[j] = f2bf(kvw[j]);
  }
}

__global__ __launch_bounds__(256) void k_prep_xT(const float* __restrict__ x,
                                                 unsigned short* __restrict__ xT) {
  int i = blockIdx.x * 256 + threadIdx.x;   // over B*NQ*CH, exact
  int c = i % CH;
  int p = (i / CH) % NQ;
  int b = i / (CH * NQ);
  xT[i] = f2bf(x[((size_t)(b * CH + c)) * NQ + p]);
}

// ---------------- depthwise pipeline ----------------
__global__ __launch_bounds__(256) void k_dw1(const float* __restrict__ x,
    const float* __restrict__ w1, const float* __restrict__ g1,
    const float* __restrict__ be1, const float* __restrict__ m1,
    const float* __restrict__ v1, const float* __restrict__ w2,
    const float* __restrict__ g2, const float* __restrict__ be2,
    const float* __restrict__ m2, const float* __restrict__ v2,
    float* __restrict__ tbuf) {
  int i = blockIdx.x * 256 + threadIdx.x;
  if (i >= BATCH * CH * NK) return;
  int pk = i % NK;
  int c  = (i / NK) % CH;
  int b  = i / (NK * CH);
  int yk = pk / WKk, xk = pk % WKk;
  float acc = 0.f;
  #pragma unroll
  for (int dy = 0; dy < 5; ++dy) {
    int y = 2 * yk - 2 + dy;
    if (y < 0 || y >= IH) continue;
    #pragma unroll
    for (int dx = 0; dx < 5; ++dx) {
      int xx = 2 * xk - 2 + dx;
      if (xx < 0 || xx >= IW) continue;
      acc += x[(((size_t)b * CH + c) * IH + y) * IW + xx] * w1[c * 25 + dy * 5 + dx];
    }
  }
  float inv1 = g1[c] * rsqrtf(v1[c] + 1e-5f);
  acc = acc * inv1 + (be1[c] - m1[c] * inv1);
  acc = fmaxf(acc, 0.f);
  acc *= w2[c];
  float inv2 = g2[c] * rsqrtf(v2[c] + 1e-5f);
  acc = acc * inv2 + (be2[c] - m2[c] * inv2);
  tbuf[i] = acc;
}

__global__ __launch_bounds__(256) void k_dw2(const float* __restrict__ tbuf,
    const float* __restrict__ lcw, const float* __restrict__ lcb,
    unsigned short* __restrict__ kvT) {
  int i = blockIdx.x * 256 + threadIdx.x;
  if (i >= BATCH * CH * NK) return;
  int pk = i % NK;
  int c  = (i / NK) % CH;
  int b  = i / (NK * CH);
  int yk = pk / WKk, xk = pk % WKk;
  const float* tc = tbuf + ((size_t)b * CH + c) * NK;
  float acc = lcb[c] + tc[pk];  // bias + residual
  #pragma unroll
  for (int dy = -1; dy <= 1; ++dy) {
    int y = yk + dy;
    if (y < 0 || y >= HKk) continue;
    #pragma unroll
    for (int dx = -1; dx <= 1; ++dx) {
      int xx = xk + dx;
      if (xx < 0 || xx >= WKk) continue;
      acc += tc[y * WKk + xx] * lcw[c * 9 + (dy + 1) * 3 + (dx + 1)];
    }
  }
  kvT[((size_t)b * NK + pk) * CH + c] = f2bf(acc);
}

// ---------------- projection GEMM: Y[co,p] = W[co,:] . X^T[p,:] + bias ----------
// One wave computes a 16(co) x 64(p) tile: 8 k-steps, 4 WMMAs each.
__global__ __launch_bounds__(256) void k_gemm(const unsigned short* __restrict__ Wb,
    const unsigned short* __restrict__ Xt, const float* __restrict__ bias,
    int Co, int P, int PP, int mode,
    unsigned short* __restrict__ qout, unsigned short* __restrict__ kout,
    unsigned short* __restrict__ vout) {
  int wid = blockIdx.x * 8 + (threadIdx.x >> 5);
  int coTiles = Co >> 4;
  int pTiles  = (PP + 63) >> 6;
  int perB    = coTiles * pTiles;
  if (wid >= BATCH * perB) return;
  int b  = wid / perB;
  int r  = wid - b * perB;
  int ct = r / pTiles;
  int pt = r - ct * pTiles;
  int co0 = ct << 4;
  int p0  = pt << 6;
  int lane = threadIdx.x & 31;
  int lm   = lane & 15;
  int hi8  = (lane & 16) ? 8 : 0;

  v8f acc[4] = {};
  const unsigned short* arow = Wb + (size_t)(co0 + lm) * CH;
  const unsigned short* brow[4];
  int pcol[4];
  #pragma unroll
  for (int t = 0; t < 4; ++t) {
    int p = p0 + t * 16 + lm;
    pcol[t] = p;
    int pc = p < P ? p : (P - 1);
    brow[t] = Xt + ((size_t)b * P + pc) * CH;
  }
  #pragma unroll
  for (int kk = 0; kk < CH / 32; ++kk) {
    v16bf av = ldop(arow + kk * 32, hi8);
    #pragma unroll
    for (int t = 0; t < 4; ++t) {
      v16bf bv = ldop(brow[t] + kk * 32, hi8);
      acc[t] = wmma_bf16(av, bv, acc[t]);
    }
  }
  #pragma unroll
  for (int t = 0; t < 4; ++t) {
    int p = pcol[t];
    #pragma unroll
    for (int i = 0; i < 8; ++i) {
      int co = co0 + i + hi8;              // C-layout: row = vgpr + 8*(lane>=16)
      float val = (p < P) ? (acc[t][i] + bias[co]) : 0.f;
      unsigned short hv = f2bf(val);
      if (mode == 0) {                      // q -> [bh, p, hd]
        if (p < P) {
          int h = co >> 5, d = co & 31;
          qout[(((size_t)(b * HEADS + h)) * NQ + p) * HD + d] = hv;
        }
      } else {
        if (co < CH) {                      // k -> [bh, key, hd]
          if (p < P) {
            int h = co >> 5, d = co & 31;
            kout[(((size_t)(b * HEADS + h)) * NK + p) * HD + d] = hv;
          }
        } else {                            // v -> transposed [bh, hd, keyPadded]
          int c2 = co - CH;
          int h = c2 >> 5, d = c2 & 31;
          if (p < PP)
            vout[(((size_t)(b * HEADS + h)) * HD + d) * NKP + p] = hv;
        }
      }
    }
  }
}

// ---------------- fused flash attention ----------------
// 128 threads / 4 waves; wave owns 16 queries; streams NK in 32-key tiles.
// K/V tiles are double-buffered in LDS via async copies (ASYNCcnt) when the
// toolchain exposes the gfx1250 async builtins; otherwise direct global loads.
__global__ __launch_bounds__(128) void k_attn(const unsigned short* __restrict__ qb,
    const unsigned short* __restrict__ kbuf, const unsigned short* __restrict__ vbuf,
    const unsigned short* __restrict__ rpe, float* __restrict__ out) {
  __shared__ unsigned short sP[4][32][16];   // per-wave P tile, [k][m]
#if HAVE_ASYNC
  __shared__ unsigned short sK[2][4][32][32]; // [buf][wave][key][hd]
  __shared__ unsigned short sV[2][4][32][32]; // [buf][wave][hd][key]
#endif
  int wv   = threadIdx.x >> 5;
  int lane = threadIdx.x & 31;
  int lm   = lane & 15;
  int hi8  = (lane & 16) ? 8 : 0;
  int bh = blockIdx.x / 49;
  int qt = blockIdx.x % 49;
  int q0 = qt * 64 + wv * 16;
  int b  = bh >> 3;
  int h  = bh & 7;

  const unsigned short* qr = qb + ((size_t)bh * NQ + q0 + lm) * HD;
  v16bf Aq = ldop(qr, hi8);

  v8f o0 = {}, o1 = {};
  float rm[8], rl[8];
  #pragma unroll
  for (int i = 0; i < 8; ++i) { rm[i] = -1e30f; rl[i] = 0.f; }

#if HAVE_ASYNC
  // Issue the 8 async B128 copies (4 for K rows, 4 for V rows) of one 32-key tile.
  auto issue_tile = [&](int kt_, int pb_) {
    int k0_ = kt_ * 32;
    int row = k0_ + lane;                       // key row for K
    row = row < NK ? row : NK - 1;
    const unsigned short* gk = kbuf + ((size_t)bh * NK + row) * HD;
    const unsigned short* gv = vbuf + ((size_t)bh * HD + lane) * NKP + k0_;
    #pragma unroll
    for (int j = 0; j < 4; ++j) {
      __builtin_amdgcn_global_load_async_to_lds_b128(
          (AS_GLOBAL v4i_t*)(gk + j * 8),
          (AS_LDS v4i_t*)&sK[pb_][wv][lane][j * 8], 0, 0);
      __builtin_amdgcn_global_load_async_to_lds_b128(
          (AS_GLOBAL v4i_t*)(gv + j * 8),
          (AS_LDS v4i_t*)&sV[pb_][wv][lane][j * 8], 0, 0);
    }
  };
  issue_tile(0, 0);
#endif

  for (int kt = 0; kt < 25; ++kt) {
    int k0   = kt * 32;
    int key0 = k0 + lm;
    int key1 = key0 + 16;
    v16bf Bk0, Bk1, Bv0, Bv1;
#if HAVE_ASYNC
    int pb = kt & 1;
    if (kt + 1 < 25) {
      issue_tile(kt + 1, pb ^ 1);   // prefetch next tile into alternate buffer
      WAIT_ASYNC(8);                // current tile's 8 copies are complete
    } else {
      WAIT_ASYNC(0);
    }
    Bk0 = ldop(&sK[pb][wv][lm][0], hi8);        // ds_load_b128 pairs
    Bk1 = ldop(&sK[pb][wv][lm + 16][0], hi8);
    Bv0 = ldop(&sV[pb][wv][lm][0], hi8);
    Bv1 = ldop(&sV[pb][wv][lm + 16][0], hi8);
#else
    int c0 = key0 < NK ? key0 : NK - 1;
    int c1 = key1 < NK ? key1 : NK - 1;
    Bk0 = ldop(kbuf + ((size_t)bh * NK + c0) * HD, hi8);
    Bk1 = ldop(kbuf + ((size_t)bh * NK + c1) * HD, hi8);
    Bv0 = ldop(vbuf + ((size_t)bh * HD + lm) * NKP + k0, hi8);
    Bv1 = ldop(vbuf + ((size_t)bh * HD + lm + 16) * NKP + k0, hi8);
    if (kt + 1 < 25) {
      int pf = (key0 + 32 < NK) ? key0 + 32 : NK - 1;
      __builtin_prefetch(kbuf + ((size_t)bh * NK + pf) * HD, 0, 0);
    }
#endif
    v8f z = {};
    v8f s0 = wmma_bf16(Aq, Bk0, z);
    v8f s1 = wmma_bf16(Aq, Bk1, z);

    float e0v[8], e1v[8];
    #pragma unroll
    for (int i = 0; i < 8; ++i) {
      int m = i + hi8;
      int q = q0 + m;
      const unsigned short* rr = rpe + ((size_t)h * NQ + q) * NK;
      float a0 = (key0 < NK) ? (s0[i] * ATT_SCALE + bf2f(rr[key0])) : -1e30f;
      float a1 = (key1 < NK) ? (s1[i] * ATT_SCALE + bf2f(rr[key1])) : -1e30f;
      // row max across the 16 lanes sharing this row (width-16 xor butterfly)
      float tv = fmaxf(a0, a1);
      tv = fmaxf(tv, __shfl_xor(tv, 1, 16));
      tv = fmaxf(tv, __shfl_xor(tv, 2, 16));
      tv = fmaxf(tv, __shfl_xor(tv, 4, 16));
      tv = fmaxf(tv, __shfl_xor(tv, 8, 16));
      float nm = fmaxf(rm[i], tv);
      float sc = __expf(rm[i] - nm);
      float e0 = __expf(a0 - nm);
      float e1 = __expf(a1 - nm);
      float ps = e0 + e1;
      ps += __shfl_xor(ps, 1, 16);
      ps += __shfl_xor(ps, 2, 16);
      ps += __shfl_xor(ps, 4, 16);
      ps += __shfl_xor(ps, 8, 16);
      rl[i] = rl[i] * sc + ps;
      rm[i] = nm;
      o0[i] *= sc;
      o1[i] *= sc;
      e0v[i] = e0;
      e1v[i] = e1;
    }
    // C-layout -> A-layout transpose of P through LDS ([k][m])
    #pragma unroll
    for (int i = 0; i < 8; ++i) {
      sP[wv][lm][i + hi8]      = f2bf(e0v[i]);
      sP[wv][16 + lm][i + hi8] = f2bf(e1v[i]);
    }
    __syncthreads();
    v16u pv;
    #pragma unroll
    for (int j = 0; j < 8; ++j) {
      pv[j]     = sP[wv][hi8 + j][lm];
      pv[j + 8] = sP[wv][hi8 + 16 + j][lm];
    }
    v16bf Ap = __builtin_bit_cast(v16bf, pv);
    o0 = wmma_bf16(Ap, Bv0, o0);
    o1 = wmma_bf16(Ap, Bv1, o1);
    __syncthreads();
  }
  #pragma unroll
  for (int i = 0; i < 8; ++i) {
    int m = i + hi8;
    int q = q0 + m;
    float inv = 1.f / rl[i];
    out[(((size_t)b * CH + h * HD + lm) * NQ) + q]        = o0[i] * inv;
    out[(((size_t)b * CH + h * HD + lm + 16) * NQ) + q]   = o1[i] * inv;
  }
}

// ---------------- launch ----------------
extern "C" void kernel_launch(void* const* d_in, const int* in_sizes, int n_in,
                              void* d_out, int out_size, void* d_ws, size_t ws_size,
                              hipStream_t stream) {
  (void)in_sizes; (void)n_in; (void)out_size; (void)ws_size;
  const float* x   = (const float*)d_in[0];
  const float* rpe = (const float*)d_in[1];
  const float* qw  = (const float*)d_in[2];
  const float* qb  = (const float*)d_in[3];
  const float* kvw = (const float*)d_in[4];
  const float* kvb = (const float*)d_in[5];
  const float* w1  = (const float*)d_in[6];
  const float* g1  = (const float*)d_in[7];
  const float* be1 = (const float*)d_in[8];
  const float* m1  = (const float*)d_in[9];
  const float* v1  = (const float*)d_in[10];
  const float* w2  = (const float*)d_in[11];
  const float* g2  = (const float*)d_in[12];
  const float* be2 = (const float*)d_in[13];
  const float* m2  = (const float*)d_in[14];
  const float* v2  = (const float*)d_in[15];
  const float* lcw = (const float*)d_in[16];
  const float* lcb = (const float*)d_in[17];

  char* ws = (char*)d_ws;
  unsigned short* rpeb = (unsigned short*)(ws + OFF_RPE);
  unsigned short* xT   = (unsigned short*)(ws + OFF_XT);
  unsigned short* wqB  = (unsigned short*)(ws + OFF_WQ);
  unsigned short* wkvB = (unsigned short*)(ws + OFF_WKV);
  float*          tbuf = (float*)(ws + OFF_T);
  unsigned short* kvT  = (unsigned short*)(ws + OFF_KVT);
  unsigned short* qbB  = (unsigned short*)(ws + OFF_Q);
  unsigned short* kB   = (unsigned short*)(ws + OFF_K);
  unsigned short* vTB  = (unsigned short*)(ws + OFF_VT);
  float* out = (float*)d_out;

  int n4 = HEADS * NQ * NK / 4;                       // 4,917,248
  k_prep_rpe<<<(n4 + 255) / 256, 256, 0, stream>>>((const float4*)rpe, (v4u*)rpeb, n4);
  k_prep_w<<<(3 * CH * CH + 255) / 256, 256, 0, stream>>>(qw, kvw, wqB, wkvB);
  k_prep_xT<<<(BATCH * NQ * CH) / 256, 256, 0, stream>>>(x, xT);
  int ndw = BATCH * CH * NK;
  k_dw1<<<(ndw + 255) / 256, 256, 0, stream>>>(x, w1, g1, be1, m1, v1, w2, g2, be2,
                                               m2, v2, tbuf);
  k_dw2<<<(ndw + 255) / 256, 256, 0, stream>>>(tbuf, lcw, lcb, kvT);

  // q projection: Co=256, P=NQ (pTiles=49) -> 6272 waves -> 784 blocks
  int wq_waves = BATCH * (CH >> 4) * ((NQ + 63) >> 6);
  k_gemm<<<(wq_waves + 7) / 8, 256, 0, stream>>>(wqB, xT, qb, CH, NQ, NQ, 0,
                                                 qbB, nullptr, nullptr);
  // kv projection: Co=512, P=NK, padded to NKP (pTiles=13) -> 3328 waves
  int wkv_waves = BATCH * ((2 * CH) >> 4) * ((NKP + 63) >> 6);
  k_gemm<<<(wkv_waves + 7) / 8, 256, 0, stream>>>(wkvB, kvT, kvb, 2 * CH, NK, NKP, 1,
                                                  nullptr, kB, vTB);

  // attention: one block per (b,h,64-query tile)
  k_attn<<<BATCH * HEADS * (NQ / 64), 128, 0, stream>>>(qbB, kB, vTB, rpeb, out);
}
